// Generator_31696858644693
// MI455X (gfx1250) — matrix operand order repaired and checked by reading, not compiled, not run on hardware
//
#include <hip/hip_runtime.h>
#include <hip/hip_bf16.h>

// Problem constants (from reference): F=32, H=128, OC=64, B=64, L=512
#define B_   64
#define L_   512
#define F_   32
#define H_   128
#define OC_  64
#define SLOPE_ 0.02f
#define BN_EPS_ 1e-5f

typedef __attribute__((ext_vector_type(16))) __bf16 v16bf;
typedef __attribute__((ext_vector_type(8)))  float  v8f;
typedef __attribute__((ext_vector_type(4)))  unsigned int v4u;

union Frag16 { v16bf v; v4u q[2]; unsigned short u[16]; };
union Acc8   { v8f   v; float f[8]; };

__device__ __forceinline__ unsigned short f2bf_bits(float f) {
  union { float f; unsigned u; } x; x.f = f;
  unsigned r = x.u + 0x7FFFu + ((x.u >> 16) & 1u);   // round-to-nearest-even
  return (unsigned short)(r >> 16);
}

// ---------------------------------------------------------------- cast f32->bf16 (plain)
__global__ void k_cast_bf16(const float* __restrict__ s, unsigned short* __restrict__ d, int n) {
  int i = blockIdx.x * blockDim.x + threadIdx.x;
  if (i < n) d[i] = f2bf_bits(s[i]);
}

// ---------------------------------------------------------------- cast f32->bf16 with (K,N)->[n*K+k] transpose
__global__ void k_cast_bf16_t(const float* __restrict__ s, unsigned short* __restrict__ d, int K, int N) {
  int i = blockIdx.x * blockDim.x + threadIdx.x;
  if (i < K * N) {
    int k = i / N, n = i - k * N;
    d[n * K + k] = f2bf_bits(s[i]);
  }
}

// ---------------------------------------------------------------- batched WMMA GEMM, fast-path only
// One wave per 64x16 output tile (4 M-subtiles reuse one B fragment).
// REQUIRES: A contiguous in K (stride am per row), B contiguous in K (stride bn per col),
//           M%64==0, N%16==0, K%32==0, am/bn multiples of 8 (16B-aligned fragments).
// C[b,m,n] = act( scale * sum_k A[b,m,k]*B[b,k,n] + bias ) -> optional f32 (cb,cm,cn) and bf16 (cb2,cm2,cn2)
__global__ __launch_bounds__(32)
void k_gemm(const unsigned short* __restrict__ A,
            const unsigned short* __restrict__ Bm,
            const float* __restrict__ bias,
            float* __restrict__ Cf, unsigned short* __restrict__ Cb,
            int K,
            long long ab, long long am,
            long long bb, long long bn,
            long long cb,  long long cm,  long long cn,
            long long cb2, long long cm2, long long cn2,
            float scale, int act, int bias_mode /*0 none,1 per-m,2 per-n*/)
{
  const int lane = threadIdx.x;
  const int half = lane >> 4;            // 0: lanes 0-15, 1: lanes 16-31
  const int low  = lane & 15;
  const int tm = blockIdx.x * 64;
  const int tn = blockIdx.y * 16;
  const long long b = blockIdx.z;

  const unsigned short* Ab = A  + b * ab;
  const unsigned short* Bb = Bm + b * bb + (long long)(tn + low) * bn;

  Acc8 acc[4] = {};

  for (int k0 = 0; k0 < K; k0 += 32) {
    // B fragment (32x16 bf16): lane holds col N=tn+low, e<8 -> K=k0+half*8+e ; e>=8 -> +16
    Frag16 fb;
    const unsigned short* pb = Bb + k0 + half * 8;
    fb.q[0] = *(const v4u*)(pb);
    fb.q[1] = *(const v4u*)(pb + 16);
    // A fragments (16x32 bf16): lane holds row M, e<8 -> K=k0+half*8+e ; e>=8 -> +16
    Frag16 fa[4];
    #pragma unroll
    for (int mt = 0; mt < 4; ++mt) {
      const unsigned short* pa = Ab + (long long)(tm + mt * 16 + low) * am + k0 + half * 8;
      fa[mt].q[0] = *(const v4u*)(pa);
      fa[mt].q[1] = *(const v4u*)(pa + 16);
    }
    if (k0 + 32 < K) {            // prefetch next K-step (global_prefetch_b8)
      __builtin_prefetch(pb + 32, 0, 1);
      __builtin_prefetch(Ab + (long long)(tm + low) * am + k0 + 32, 0, 1);
    }
    #pragma unroll
    for (int mt = 0; mt < 4; ++mt)
      acc[mt].v = __builtin_amdgcn_wmma_f32_16x16x32_bf16(false, fa[mt].v, false, fb.v,
                                                          (short)0, acc[mt].v, false, false);
  }

  const int n = tn + low;
  #pragma unroll
  for (int mt = 0; mt < 4; ++mt) {
    #pragma unroll
    for (int r = 0; r < 8; ++r) {
      const int m = tm + mt * 16 + r + half * 8;   // C: VGPR r -> M=r (lanes 0-15) / M=8+r (lanes 16-31)
      float v = acc[mt].f[r] * scale;
      if (bias_mode == 1)      v += bias[m];
      else if (bias_mode == 2) v += bias[n];
      if (act == 1) v = (v > 0.f) ? v : v * SLOPE_;
      if (Cf) Cf[b * cb  + (long long)m * cm  + (long long)n * cn ] = v;
      if (Cb) Cb[b * cb2 + (long long)m * cm2 + (long long)n * cn2] = f2bf_bits(v);
    }
  }
}

// ---------------------------------------------------------------- BN stats (per channel over B,L)
__global__ void k_bn_stats(const float* __restrict__ wy,
                           const float* __restrict__ bn_g, const float* __restrict__ bn_b,
                           float* __restrict__ scale, float* __restrict__ shift)
{
  __shared__ float s_sum[256], s_sq[256];
  const int c = blockIdx.x, tid = threadIdx.x;
  float sum = 0.f, sq = 0.f;
  for (int i = tid; i < B_ * L_; i += 256) {
    int b = i >> 9, l = i & (L_ - 1);
    float v = wy[(long long)b * H_ * L_ + (long long)c * L_ + l];
    sum += v; sq += v * v;
  }
  s_sum[tid] = sum; s_sq[tid] = sq; __syncthreads();
  for (int s = 128; s > 0; s >>= 1) {
    if (tid < s) { s_sum[tid] += s_sum[tid + s]; s_sq[tid] += s_sq[tid + s]; }
    __syncthreads();
  }
  if (tid == 0) {
    const float inv = 1.0f / (float)(B_ * L_);
    float mu  = s_sum[0] * inv;
    float var = s_sq[0] * inv - mu * mu;
    float sc  = bn_g[c] * rsqrtf(var + BN_EPS_);
    scale[c] = sc;
    shift[c] = bn_b[c] - mu * sc;
  }
}

// ---------------------------------------------------------------- BN apply + residual(s)
// f32 out in (B,H,L); bf16 out in token-major (B,L,H)
__global__ void k_bn_res(const float* __restrict__ wy,
                         const float* __restrict__ scale, const float* __restrict__ shift,
                         const float* __restrict__ res1, const float* __restrict__ res2,
                         float* __restrict__ outf, unsigned short* __restrict__ outb, int n)
{
  int i = blockIdx.x * blockDim.x + threadIdx.x;
  if (i >= n) return;
  int b   = i >> 16;                    // H*L == 65536
  int rem = i & 65535;
  int c   = rem >> 9;                   // / L
  int l   = rem & (L_ - 1);
  float v = wy[i] * scale[c] + shift[c] + res1[i];
  if (res2) v += res2[i];
  outf[i] = v;
  outb[((long long)b * L_ + l) * H_ + c] = f2bf_bits(v);
}

// ---------------------------------------------------------------- fc6 + sigmoid + fixup + concat
__global__ void k_final(const float* __restrict__ x, const float* __restrict__ d2,
                        const float* __restrict__ w6, const float* __restrict__ b6,
                        float* __restrict__ out)
{
  int t = blockIdx.x * blockDim.x + threadIdx.x;
  if (t >= B_ * L_) return;
  const float* xr = x  + (long long)t * F_;
  const float* hr = d2 + (long long)t * F_;
  float geo[4];
  #pragma unroll
  for (int j = 0; j < 4; ++j) {
    float acc = b6[j];
    for (int k = 0; k < F_; ++k) acc += hr[k] * w6[k * 4 + j];
    geo[j] = 1.0f / (1.0f + expf(-acc));
  }
  float a1 = xr[F_ - 1];
  geo[3] = (a1 == 0.0f ? 1.0f : 0.0f) * (geo[3] - geo[1]) + a1 * (geo[2] - geo[0]) + geo[1];
  float* o = out + (long long)t * 30;
  for (int k = 0; k < 26; ++k) o[k] = xr[k];
  #pragma unroll
  for (int j = 0; j < 4; ++j) o[26 + j] = geo[j];
}

// ================================================================ host
extern "C" void kernel_launch(void* const* d_in, const int* in_sizes, int n_in,
                              void* d_out, int out_size, void* d_ws, size_t ws_size,
                              hipStream_t stream)
{
  (void)in_sizes; (void)n_in; (void)out_size; (void)ws_size;
  // Input order: [0]=x ; params leaves in jax pytree order (sorted dict keys):
  // attn[a] (base=1+10a): W_b,W_w,bn_b,bn_g,g_b,g_w,ph_b,ph_w,th_b,th_w
  // dec0: b@41,w@42 ; dec1: b@43,w@44 ; enc0: b@45,w@46 ; enc1: b@47,w@48 ; enc2: b@49,w@50 ; fc6: b@51,w@52
  const float* xin = (const float*)d_in[0];

  char* ws = (char*)d_ws;
  size_t off = 0;
  auto alloc = [&](size_t bytes) -> void* {
    off = (off + 255) & ~(size_t)255;
    void* p = ws + off; off += bytes; return p;
  };
  const long long HL = (long long)H_ * L_;

  unsigned short* xb     = (unsigned short*)alloc((size_t)B_ * L_ * F_ * 2);
  unsigned short* w_enc0 = (unsigned short*)alloc((size_t)F_ * 2 * F_ * 2);   // transposed [n*K+k]
  unsigned short* w_enc1 = (unsigned short*)alloc((size_t)2 * F_ * H_ * 2);   // transposed
  unsigned short* w_enc2 = (unsigned short*)alloc((size_t)H_ * H_ * 2);       // transposed
  unsigned short *w_gw[4], *w_thw[4], *w_phw[4], *w_Ww[4];
  for (int a = 0; a < 4; ++a) {
    w_Ww[a]  = (unsigned short*)alloc((size_t)H_ * OC_ * 2);   // plain (A operand, k contiguous)
    w_gw[a]  = (unsigned short*)alloc((size_t)OC_ * H_ * 2);   // plain (B operand, k contiguous)
    w_phw[a] = (unsigned short*)alloc((size_t)OC_ * H_ * 2);   // plain
    w_thw[a] = (unsigned short*)alloc((size_t)OC_ * H_ * 2);   // plain
  }
  unsigned short* w_dec0 = (unsigned short*)alloc((size_t)H_ * 64 * 2);       // transposed
  unsigned short* w_dec1 = (unsigned short*)alloc((size_t)64 * F_ * 2);       // transposed
  unsigned short* h1b    = (unsigned short*)alloc((size_t)B_ * L_ * 64 * 2);
  unsigned short* h2b    = (unsigned short*)alloc((size_t)B_ * L_ * H_ * 2);
  float* A0f = (float*)alloc((size_t)B_ * HL * 4);   // f32 activations, (B,H,L)
  float* A1f = (float*)alloc((size_t)B_ * HL * 4);
  float* A2f = (float*)alloc((size_t)B_ * HL * 4);
  unsigned short* B0b = (unsigned short*)alloc((size_t)B_ * HL * 2);  // bf16 activations, token-major (B,L,H)
  unsigned short* B1b = (unsigned short*)alloc((size_t)B_ * HL * 2);
  unsigned short* gbuf  = (unsigned short*)alloc((size_t)B_ * OC_ * L_ * 2);  // g^T  (B,OC,L)
  unsigned short* thbuf = (unsigned short*)alloc((size_t)B_ * L_ * OC_ * 2);  // th   (B,L,OC)
  unsigned short* phbuf = (unsigned short*)alloc((size_t)B_ * OC_ * L_ * 2);  // ph   (B,OC,L)
  unsigned short* ybuf  = (unsigned short*)alloc((size_t)B_ * L_ * OC_ * 2);  // y    (B,L,OC)
  unsigned short* Sbuf  = (unsigned short*)alloc((size_t)B_ * OC_ * OC_ * 2); // S^T  (B,OC,OC)
  float* wyf = (float*)alloc((size_t)B_ * HL * 4);
  float* scl = (float*)alloc((size_t)H_ * 4);
  float* shf = (float*)alloc((size_t)H_ * 4);
  unsigned short* d1b = (unsigned short*)alloc((size_t)B_ * L_ * 64 * 2);
  float* d2f = (float*)alloc((size_t)B_ * L_ * F_ * 4);

  auto CAST = [&](const void* src, unsigned short* dst, int n) {
    k_cast_bf16<<<dim3((n + 255) / 256), dim3(256), 0, stream>>>((const float*)src, dst, n);
  };
  auto CAST_T = [&](const void* src, unsigned short* dst, int K, int N) {
    int n = K * N;
    k_cast_bf16_t<<<dim3((n + 255) / 256), dim3(256), 0, stream>>>((const float*)src, dst, K, N);
  };
  auto GEMM = [&](const unsigned short* A, const unsigned short* Bm, const float* bias, int bias_mode,
                  float* Cf, unsigned short* Cb, int M, int N, int K,
                  long long ab, long long am,
                  long long bb, long long bn,
                  long long cb,  long long cm,  long long cn,
                  long long cb2, long long cm2, long long cn2,
                  float scale, int act) {
    dim3 g(M / 64, N / 16, B_);
    k_gemm<<<g, dim3(32), 0, stream>>>(A, Bm, bias, Cf, Cb, K,
                                       ab, am, bb, bn, cb, cm, cn, cb2, cm2, cn2,
                                       scale, act, bias_mode);
  };

  // ---- casts (weights + input)
  CAST(d_in[0],  xb, B_ * L_ * F_);
  CAST_T(d_in[46], w_enc0, F_,     2 * F_);
  CAST_T(d_in[48], w_enc1, 2 * F_, H_);
  CAST_T(d_in[50], w_enc2, H_,     H_);
  for (int a = 0; a < 4; ++a) {
    int base = 1 + a * 10;
    CAST(d_in[base + 1], w_Ww[a],  H_ * OC_);
    CAST(d_in[base + 5], w_gw[a],  OC_ * H_);
    CAST(d_in[base + 7], w_phw[a], OC_ * H_);
    CAST(d_in[base + 9], w_thw[a], OC_ * H_);
  }
  CAST_T(d_in[42], w_dec0, H_, 64);
  CAST_T(d_in[44], w_dec1, 64, F_);

  // ---- encoder (tokens; all operands contiguous-K)
  GEMM(xb,  w_enc0, (const float*)d_in[45], 2, nullptr, h1b, L_, 2 * F_, F_,
       (long long)L_ * F_, F_,   0, F_,
       0, 0, 0,   (long long)L_ * 64, 64, 1,  1.f, 1);
  GEMM(h1b, w_enc1, (const float*)d_in[47], 2, nullptr, h2b, L_, H_, 2 * F_,
       (long long)L_ * 64, 64,   0, 64,
       0, 0, 0,   (long long)L_ * H_, H_, 1,  1.f, 1);
  // enc2: f32 -> (B,H,L), bf16 -> token-major (B,L,H)
  GEMM(h2b, w_enc2, (const float*)d_in[49], 2, A0f, B0b, L_, H_, H_,
       (long long)L_ * H_, H_,   0, H_,
       HL, 1, L_,   HL, H_, 1,  1.f, 0);

  // ---- attention block: y = th @ ((ph@g)/L) ; wy = W_w @ y^T ; BN over (B,L); + residuals
  auto ATTN = [&](int a, const float* Xinf, const unsigned short* Xinb,
                  float* Xoutf, unsigned short* Xoutb, const float* res2) {
    int base = 1 + a * 10;
    const float* W_b  = (const float*)d_in[base + 0];
    const float* bn_b = (const float*)d_in[base + 2];
    const float* bn_g = (const float*)d_in[base + 3];
    const float* g_b  = (const float*)d_in[base + 4];
    const float* ph_b = (const float*)d_in[base + 6];
    const float* th_b = (const float*)d_in[base + 8];

    // g^T (B,OC,L): C[m=l,n=o] stored at [o*L+l]
    GEMM(Xinb, w_gw[a], g_b, 2, nullptr, gbuf, L_, OC_, H_,
         HL, H_,   0, H_,
         0, 0, 0,   (long long)OC_ * L_, 1, L_,  1.f, 0);
    // th (B,L,OC)
    GEMM(Xinb, w_thw[a], th_b, 2, nullptr, thbuf, L_, OC_, H_,
         HL, H_,   0, H_,
         0, 0, 0,   (long long)L_ * OC_, OC_, 1,  1.f, 0);
    // ph^T == (B,OC,L): ph[o,l] = sum_c X[c,l] ph_w[o,c] + ph_b[o]
    GEMM(Xinb, w_phw[a], ph_b, 2, nullptr, phbuf, L_, OC_, H_,
         HL, H_,   0, H_,
         0, 0, 0,   (long long)OC_ * L_, 1, L_,  1.f, 0);
    // S^T (B,OC,OC): S = (ph @ g)/L ; A=ph (B,OC,L), B=g^T (k=l contiguous)
    GEMM(phbuf, gbuf, nullptr, 0, nullptr, Sbuf, OC_, OC_, L_,
         (long long)OC_ * L_, L_,   (long long)OC_ * L_, L_,
         0, 0, 0,   (long long)OC_ * OC_, 1, OC_,  1.f / (float)L_, 0);
    // y (B,L,OC) = th @ S ; B[k=o,n=o'] = S^T[o'*OC+o] (k contiguous)
    GEMM(thbuf, Sbuf, nullptr, 0, nullptr, ybuf, L_, OC_, OC_,
         (long long)L_ * OC_, OC_,   (long long)OC_ * OC_, OC_,
         0, 0, 0,   (long long)L_ * OC_, OC_, 1,  1.f, 0);
    // wy (B,H,L) f32 = W_w @ y^T + W_b ; B[k=o,n=l] = y[l*OC+o] (k contiguous)
    GEMM(w_Ww[a], ybuf, W_b, 1, wyf, nullptr, H_, L_, OC_,
         0, OC_,   (long long)L_ * OC_, OC_,
         HL, L_, 1,   0, 0, 0,  1.f, 0);
    // BN stats + apply + residual(s)
    k_bn_stats<<<dim3(H_), dim3(256), 0, stream>>>(wyf, bn_g, bn_b, scl, shf);
    int n = B_ * H_ * L_;
    k_bn_res<<<dim3(n / 256), dim3(256), 0, stream>>>(wyf, scl, shf, Xinf, res2, Xoutf, Xoutb, n);
  };

  ATTN(0, A0f, B0b, A1f, B1b, nullptr);  // h = attn0(enc)
  ATTN(1, A1f, B1b, A2f, B0b, A0f);      // h = attn1(h) + xhat(enc)
  ATTN(2, A2f, B0b, A1f, B1b, nullptr);  // h = attn2(h)
  ATTN(3, A1f, B1b, A0f, B0b, A2f);      // h = attn3(h) + xhat2

  // ---- decoder (tokens; A = token-major bf16 activations)
  GEMM(B0b, w_dec0, (const float*)d_in[41], 2, nullptr, d1b, L_, 64, H_,
       HL, H_,   0, H_,
       0, 0, 0,   (long long)L_ * 64, 64, 1,  1.f, 1);
  GEMM(d1b, w_dec1, (const float*)d_in[43], 2, d2f, nullptr, L_, F_, 64,
       (long long)L_ * 64, 64,   0, 64,
       (long long)L_ * F_, F_, 1,   0, 0, 0,  1.f, 0);

  // ---- fc6 + sigmoid + fixup + concat
  k_final<<<dim3((B_ * L_) / 256), dim3(256), 0, stream>>>(
      xin, d2f, (const float*)d_in[52], (const float*)d_in[51], (float*)d_out);
}
